// Block_35132832481325
// MI455X (gfx1250) — matrix operand order repaired and checked by reading, not compiled, not run on hardware
//
#include <hip/hip_runtime.h>
#include <hip/hip_bf16.h>
#include <math.h>

// ---------------------------------------------------------------------------
// EViT block for MI455X (gfx1250, wave32). All GEMMs via v_wmma_f32_16x16x32_bf16
// (bf16 operands, f32 accumulate). LN / softmax / top-k / GELU in fp32.
// B operands (weights, V) are stored pre-transposed so every WMMA B-fragment is
// a K-contiguous 128-bit load; A tiles are double-buffered in LDS with
// global_load_async_to_lds_b128 (ASYNCcnt path) overlapping the WMMAs.
// ---------------------------------------------------------------------------

typedef __attribute__((ext_vector_type(16))) __bf16 bf16x16;
typedef __attribute__((ext_vector_type(8)))  float  f32x8;

#define NTOK   197
#define NP     224     // padded tokens (multiple of 32)
#define NHEAD  12
#define HD     64
#define CDIM   768
#define MLPD   3072
#define NKEEP  138
#define NKEEP1 139
#define MP     144     // padded kept tokens (multiple of 16)
#define NPATCH 196
#define NKATTN 97

__device__ __forceinline__ unsigned short f2bf(float x) {
  unsigned u = __float_as_uint(x);
  unsigned r = u + 0x7fffu + ((u >> 16) & 1u);
  return (unsigned short)(r >> 16);
}

// A-fragment (16x32 bf16). Lane half h: K in {16h..16h+15},
// VGPR j -> pair K=16h+2j. rowptr = row base
// (row-major, K contiguous), must be 4B aligned.
__device__ __forceinline__ bf16x16 frag_a(const unsigned short* rowptr, int half) {
  union { unsigned u[8]; bf16x16 v; } f;
  const unsigned* p = (const unsigned*)rowptr;
  int b = half * 8;
#pragma unroll
  for (int j = 0; j < 4; ++j) { f.u[j] = p[b + j]; f.u[4 + j] = p[b + 4 + j]; }
  return f.v;
}

// B-fragment (32x16 bf16). Lane = column n, half h holds K=16h..16h+15
// (8 contiguous dwords). colptr = base of 32 contiguous K-elements for this
// column (column of B stored K-contiguous), 16B aligned -> b128 loads.
__device__ __forceinline__ bf16x16 frag_b(const unsigned short* colptr, int half) {
  union { unsigned u[8]; bf16x16 v; } f;
  const unsigned* p = (const unsigned*)colptr;
  int b = half * 8;
#pragma unroll
  for (int j = 0; j < 8; ++j) f.u[j] = p[b + j];
  return f.v;
}

__device__ __forceinline__ f32x8 wmma_bf16(bf16x16 a, bf16x16 b, f32x8 c) {
  return __builtin_amdgcn_wmma_f32_16x16x32_bf16(false, a, false, b, (short)0, c, false, false);
}

// Async 16B lane-copy global -> LDS (CDNA5 async path, tracked by ASYNCcnt).
// lds_off: byte offset inside the kernel's (single) LDS allocation.
__device__ __forceinline__ void async_cp16(unsigned lds_off, const void* gaddr) {
  asm volatile("global_load_async_to_lds_b128 %0, %1, off"
               :: "v"(lds_off), "v"(gaddr) : "memory");
}
template <int N>
__device__ __forceinline__ void wait_async() {
  asm volatile("s_wait_asynccnt %0" :: "i"(N) : "memory");
}

// ---------------------------------------------------------------------------
// Generic tiled GEMM: C = A(bf16, MxK rm) * Bt(bf16, NxK rm == B^T) + epilogue.
// Block tile 128x128, 8 waves (2x4), wave tile 64x32 = 4x2 WMMA tiles.
// A tile (128x32, 8KB) double-buffered in LDS via async-to-LDS; B direct (L2).
// ---------------------------------------------------------------------------
enum { EPI_QKV = 0, EPI_PROJ = 1, EPI_FC1 = 2, EPI_FC2 = 3 };

template <int EPI>
__global__ __launch_bounds__(256) void gemm_bf16_kernel(
    const unsigned short* __restrict__ A,
    const unsigned short* __restrict__ Bt,
    const float* __restrict__ bias,
    const float* __restrict__ resid,
    void* __restrict__ outp,
    int M, int N, int K) {
  __shared__ __align__(16) unsigned short la[2][128 * 32];  // [buf][m][k]
  const int tid = threadIdx.x;
  const int m0 = blockIdx.y * 128, n0 = blockIdx.x * 128;
  const int wave = __builtin_amdgcn_readfirstlane(tid >> 5);  // scalar wave id
  const int lane = tid & 31, half = lane >> 4, l16 = lane & 15;
  const int wr0 = (wave & 1) * 64, wc0 = (wave >> 1) * 32;

  // B columns this lane owns (K-contiguous rows of B^T)
  const unsigned short* bcol[2];
#pragma unroll
  for (int j = 0; j < 2; ++j)
    bcol[j] = Bt + (size_t)(n0 + wc0 + j * 16 + l16) * K;

  f32x8 acc[4][2];
#pragma unroll
  for (int i = 0; i < 4; ++i)
#pragma unroll
    for (int j = 0; j < 2; ++j)
#pragma unroll
      for (int r = 0; r < 8; ++r) acc[i][j][r] = 0.f;

  const unsigned o1 = tid * 16u, o2 = 4096u + tid * 16u;  // byte offsets in tile
  const char* Ab = (const char*)A;
  auto issue_tile = [&](int k0, int buf) {
    const char* g1 = Ab + ((size_t)(m0 + (int)(o1 >> 6)) * K + k0) * 2 + (o1 & 63u);
    const char* g2 = Ab + ((size_t)(m0 + (int)(o2 >> 6)) * K + k0) * 2 + (o2 & 63u);
    unsigned base = (unsigned)buf * 8192u;
    async_cp16(base + o1, g1);
    async_cp16(base + o2, g2);
  };

  const int nk = K >> 5;
  issue_tile(0, 0);
  for (int i = 0; i < nk; ++i) {
    const int k0 = i << 5;
    if (i + 1 < nk) {
      issue_tile(k0 + 32, (i + 1) & 1);   // stream next tile into other buffer
      wait_async<2>();                    // our 2 copies of tile i are done
      __builtin_prefetch(bcol[0] + k0 + 32, 0, 3);
      __builtin_prefetch(bcol[1] + k0 + 32, 0, 3);
    } else {
      wait_async<0>();
    }
    __syncthreads();

    const unsigned short* lab = la[i & 1];
    bf16x16 af[4], bfv[2];
#pragma unroll
    for (int t = 0; t < 4; ++t) af[t] = frag_a(lab + (wr0 + t * 16 + l16) * 32, half);
#pragma unroll
    for (int j = 0; j < 2; ++j) bfv[j] = frag_b(bcol[j] + k0, half);
#pragma unroll
    for (int t = 0; t < 4; ++t)
#pragma unroll
      for (int j = 0; j < 2; ++j)
        acc[t][j] = wmma_bf16(af[t], bfv[j], acc[t][j]);
    __syncthreads();  // tile (i&1) free for reuse at iteration i+2
  }

  // Epilogue. C/D layout: lane n=l16, VGPR r -> M = r + 8*half.
#pragma unroll
  for (int i = 0; i < 4; ++i) {
#pragma unroll
    for (int j = 0; j < 2; ++j) {
      int gn = n0 + wc0 + j * 16 + l16;
      int gmb = m0 + wr0 + i * 16 + half * 8;
#pragma unroll
      for (int r = 0; r < 8; ++r) {
        int gm = gmb + r;
        float v = acc[i][j][r] + bias[gn];
        if (EPI == EPI_QKV) {
          // q,k -> [token][d] (q scaled by D^-0.5); v -> transposed [d][token]
          int bidx = gm / NP, n = gm - bidx * NP;
          int which = gn / CDIM, rem = gn - which * CDIM;
          int head = rem >> 6, dd = rem & 63;
          size_t base = (size_t)(bidx * NHEAD + head) * 3 * NP * HD;
          unsigned short* q = (unsigned short*)outp;
          if (which == 0)      q[base + (size_t)n * HD + dd] = f2bf(v * 0.125f);
          else if (which == 1) q[base + NP * HD + (size_t)n * HD + dd] = f2bf(v);
          else                 q[base + 2 * NP * HD + (size_t)dd * NP + n] = f2bf(v);
        } else if (EPI == EPI_PROJ) {
          int bidx = gm / NP, n = gm - bidx * NP;
          if (n < NTOK) {
            size_t o = ((size_t)bidx * NTOK + n) * CDIM + gn;
            ((float*)outp)[o] = v + resid[o];
          }
        } else if (EPI == EPI_FC1) {
          float g = 0.5f * v * (1.0f + erff(v * 0.70710678f));  // exact GELU
          ((unsigned short*)outp)[(size_t)gm * MLPD + gn] = f2bf(g);
        } else {  // EPI_FC2
          int bidx = gm / MP, n = gm - bidx * MP;
          if (n < NKEEP1) {
            size_t o = ((size_t)bidx * NKEEP1 + n) * CDIM + gn;
            ((float*)outp)[o] = v + resid[o];
          }
        }
      }
    }
  }
}

// ---------------------------------------------------------------------------
// Attention: per (b,h, 16-row tile): S = q*K^T (WMMA), softmax+eps+mask in f32,
// head-pooled cls/patch atomics, then attn*V^T (WMMA) -> x_att bf16.
// ---------------------------------------------------------------------------
__global__ __launch_bounds__(256) void attn_kernel(
    const unsigned short* __restrict__ qkvs, const float* __restrict__ amask,
    unsigned short* __restrict__ xatt, float* __restrict__ clsw,
    float* __restrict__ patch) {
  __shared__ __align__(16) float ssc[16 * NP];
  __shared__ __align__(16) unsigned short sat[16 * NP];
  const int rt = blockIdx.x, bh = blockIdx.y;
  const int b = bh / NHEAD, h = bh - b * NHEAD;
  const int tid = threadIdx.x;
  const int wave = __builtin_amdgcn_readfirstlane(tid >> 5);  // scalar wave id
  const int lane = tid & 31, half = lane >> 4, l16 = lane & 15;
  const unsigned short* qb = qkvs + (size_t)bh * 3 * NP * HD;
  const unsigned short* kb = qb + NP * HD;        // [token][d]
  const unsigned short* vb = qb + 2 * NP * HD;    // transposed: [d][token]

  // -- scores: 14 column tiles over 8 waves (scalar loop, EXEC stays all-1s)
  for (int ct = wave; ct < NP / 16; ct += 8) {
    f32x8 acc;
#pragma unroll
    for (int r = 0; r < 8; ++r) acc[r] = 0.f;
#pragma unroll
    for (int kc = 0; kc < HD; kc += 32) {
      bf16x16 a = frag_a(qb + (rt * 16 + l16) * HD + kc, half);
      bf16x16 bb = frag_b(kb + (ct * 16 + l16) * HD + kc, half);  // B[k][n]=K[n][k]
      acc = wmma_bf16(a, bb, acc);
    }
#pragma unroll
    for (int r = 0; r < 8; ++r) ssc[(half * 8 + r) * NP + ct * 16 + l16] = acc[r];
  }
  __syncthreads();

  // -- softmax (row = 16 threads; shfl-xor width 16 reductions)
  {
    int row = tid >> 4, c16 = tid & 15;
    int grow = rt * 16 + row;
    float* srow = ssc + row * NP;
    unsigned short* arow = sat + row * NP;
    if (grow < NTOK) {
      float mx = -1e30f;
      for (int c = c16; c < NTOK; c += 16) mx = fmaxf(mx, srow[c]);
      for (int o = 8; o; o >>= 1) mx = fmaxf(mx, __shfl_xor(mx, o, 16));
      const float* mr = amask + ((size_t)b * NTOK + grow) * NTOK;
      float sum = 0.f;
      for (int c = c16; c < NTOK; c += 16) {
        float e = __expf(srow[c] - mx) * mr[c];
        srow[c] = e; sum += e;
      }
      for (int o = 8; o; o >>= 1) sum += __shfl_xor(sum, o, 16);
      float inv = 1.0f / (sum + 1e-6f);
      const float eN = 1e-6f / 197.0f;
      for (int c = c16; c < NP; c += 16) {
        float a = (c < NTOK) ? (srow[c] + eN) * inv : 0.f;
        arow[c] = f2bf(a);
        if (c >= 1 && c < NTOK) {
          if (grow == 0)
            atomicAdd(clsw + (size_t)b * NPATCH + (c - 1), a * (1.0f / 12.0f));
          else
            atomicAdd(patch + ((size_t)b * NPATCH + (grow - 1)) * NPATCH + (c - 1), a * (1.0f / 12.0f));
        }
      }
    } else {
      for (int c = c16; c < NP; c += 16) arow[c] = 0;
    }
  }
  __syncthreads();

  // -- attn @ V : waves 0..3 each own one 16-wide D tile; K loop over 224 keys.
  // V^T layout makes each B-fragment two contiguous b128 loads per column.
  if (wave < 4) {
    int col = wave * 16 + l16;
    const unsigned short* vcol = vb + (size_t)col * NP;
    f32x8 acc;
#pragma unroll
    for (int r = 0; r < 8; ++r) acc[r] = 0.f;
#pragma unroll
    for (int kc = 0; kc < NP; kc += 32) {
      bf16x16 a = frag_a(sat + l16 * NP + kc, half);
      bf16x16 bb = frag_b(vcol + kc, half);
      acc = wmma_bf16(a, bb, acc);
    }
#pragma unroll
    for (int r = 0; r < 8; ++r) {
      int rr = rt * 16 + half * 8 + r;
      xatt[((size_t)b * NP + rr) * CDIM + h * HD + col] = f2bf(acc[r]);
    }
  }
}

// ---------------------------------------------------------------------------
// LayerNorm1: x[B,197,C] -> bf16 h[B,224,C], zero-padded rows.
// ---------------------------------------------------------------------------
__global__ __launch_bounds__(256) void ln1_kernel(
    const float* __restrict__ x, const float* __restrict__ g,
    const float* __restrict__ be, unsigned short* __restrict__ h) {
  __shared__ float red[256];
  int blk = blockIdx.x, tid = threadIdx.x;
  int b = blk / NP, n = blk - b * NP;
  unsigned short* hrow = h + (size_t)blk * CDIM;
  if (n >= NTOK) { for (int c = tid; c < CDIM; c += 256) hrow[c] = 0; return; }
  const float* xr = x + ((size_t)b * NTOK + n) * CDIM;
  float s = 0.f, s2 = 0.f;
  for (int c = tid; c < CDIM; c += 256) { float v = xr[c]; s += v; s2 += v * v; }
  red[tid] = s; __syncthreads();
  for (int st = 128; st; st >>= 1) { if (tid < st) red[tid] += red[tid + st]; __syncthreads(); }
  float mu = red[0] * (1.f / CDIM); __syncthreads();
  red[tid] = s2; __syncthreads();
  for (int st = 128; st; st >>= 1) { if (tid < st) red[tid] += red[tid + st]; __syncthreads(); }
  float var = red[0] * (1.f / CDIM) - mu * mu;
  float rstd = rsqrtf(var + 1e-5f);
  for (int c = tid; c < CDIM; c += 256)
    hrow[c] = f2bf((xr[c] - mu) * rstd * g[c] + be[c]);
}

// ---------------------------------------------------------------------------
// Gather pruned tokens + LayerNorm2: x1[B,197,C] -> x2[B,139,C] f32 + h2 bf16.
// ---------------------------------------------------------------------------
__global__ __launch_bounds__(256) void gather_ln2_kernel(
    const float* __restrict__ x1, const int* __restrict__ tok,
    const float* __restrict__ g, const float* __restrict__ be,
    float* __restrict__ x2, unsigned short* __restrict__ h2) {
  __shared__ float red[256];
  int blk = blockIdx.x, tid = threadIdx.x;
  int b = blk / MP, n = blk - b * MP;
  unsigned short* hrow = h2 + (size_t)blk * CDIM;
  if (n >= NKEEP1) { for (int c = tid; c < CDIM; c += 256) hrow[c] = 0; return; }
  int srow = (n == 0) ? 0 : (1 + tok[b * NKEEP + n - 1]);
  const float* xr = x1 + ((size_t)b * NTOK + srow) * CDIM;
  float* x2r = x2 + ((size_t)b * NKEEP1 + n) * CDIM;
  float s = 0.f, s2 = 0.f;
  for (int c = tid; c < CDIM; c += 256) { float v = xr[c]; s += v; s2 += v * v; }
  red[tid] = s; __syncthreads();
  for (int st = 128; st; st >>= 1) { if (tid < st) red[tid] += red[tid + st]; __syncthreads(); }
  float mu = red[0] * (1.f / CDIM); __syncthreads();
  red[tid] = s2; __syncthreads();
  for (int st = 128; st; st >>= 1) { if (tid < st) red[tid] += red[tid + st]; __syncthreads(); }
  float var = red[0] * (1.f / CDIM) - mu * mu;
  float rstd = rsqrtf(var + 1e-5f);
  for (int c = tid; c < CDIM; c += 256) {
    float v = xr[c];
    x2r[c] = v;
    hrow[c] = f2bf((v - mu) * rstd * g[c] + be[c]);
  }
}

// ---------------------------------------------------------------------------
// Top-138 of cls attention per batch (stable ties -> lowest index, like top_k).
// ---------------------------------------------------------------------------
__global__ __launch_bounds__(256) void topk_kernel(
    const float* __restrict__ clsw, int* __restrict__ tokw,
    int* __restrict__ tok_out, float* __restrict__ cls_out) {
  __shared__ float sv[256]; __shared__ float wv[256]; __shared__ int wi[256];
  int b = blockIdx.x, tid = threadIdx.x;
  float v = (tid < NPATCH) ? clsw[b * NPATCH + tid] : -1e30f;
  if (tid < NPATCH) cls_out[b * NPATCH + tid] = v;
  sv[tid] = v; __syncthreads();
  for (int t = 0; t < NKEEP; ++t) {
    wv[tid] = sv[tid]; wi[tid] = tid; __syncthreads();
    for (int st = 128; st; st >>= 1) {
      if (tid < st) {
        if (wv[tid + st] > wv[tid] ||
            (wv[tid + st] == wv[tid] && wi[tid + st] < wi[tid])) {
          wv[tid] = wv[tid + st]; wi[tid] = wi[tid + st];
        }
      }
      __syncthreads();
    }
    if (tid == 0) {
      int id = wi[0];
      tokw[b * NKEEP + t] = id;
      tok_out[b * NKEEP + t] = id;
      sv[id] = -1e30f;
    }
    __syncthreads();
  }
}

// ---------------------------------------------------------------------------
// Attention-pruning mask: per (b, kept row i): gather pp row, 97th-largest
// threshold via repeated argmax, write interior of new_mask.
// ---------------------------------------------------------------------------
__global__ __launch_bounds__(256) void maskk_kernel(
    const float* __restrict__ patch, const int* __restrict__ tok,
    float* __restrict__ maskout) {
  __shared__ int st[NKEEP]; __shared__ float sp[NKEEP]; __shared__ float sw[256];
  __shared__ float wv[256]; __shared__ int wi[256]; __shared__ float sthr;
  int i = blockIdx.x, b = blockIdx.y, tid = threadIdx.x;
  for (int j = tid; j < NKEEP; j += 256) st[j] = tok[b * NKEEP + j];
  __syncthreads();
  const float* prow = patch + ((size_t)b * NPATCH + st[i]) * NPATCH;
  for (int j = tid; j < NKEEP; j += 256) sp[j] = prow[st[j]];
  __syncthreads();
  sw[tid] = (tid < NKEEP) ? sp[tid] : -1e30f;
  __syncthreads();
  for (int t = 0; t < NKATTN; ++t) {
    wv[tid] = sw[tid]; wi[tid] = tid; __syncthreads();
    for (int s2 = 128; s2; s2 >>= 1) {
      if (tid < s2) {
        if (wv[tid + s2] > wv[tid] ||
            (wv[tid + s2] == wv[tid] && wi[tid + s2] < wi[tid])) {
          wv[tid] = wv[tid + s2]; wi[tid] = wi[tid + s2];
        }
      }
      __syncthreads();
    }
    if (tid == 0) { sthr = wv[0]; sw[wi[0]] = -1e30f; }
    __syncthreads();
  }
  float thr = sthr;
  float* mrow = maskout + ((size_t)b * NKEEP1 + 1 + i) * NKEEP1;
  for (int j = tid; j < NKEEP; j += 256) mrow[1 + j] = (sp[j] >= thr) ? 1.f : 0.f;
}

// ---------------------------------------------------------------------------
__global__ void fill_kernel(float* p, float v, int n) {
  int i = blockIdx.x * 256 + threadIdx.x;
  if (i < n) p[i] = v;
}
// f32 weight (K x N, row-major) -> bf16 transposed (N x K, row-major)
__global__ void cvt_t_kernel(const float* s, unsigned short* d, int K, int N) {
  int i = blockIdx.x * 256 + threadIdx.x;
  if (i < K * N) {
    int n = i / K, k = i - n * K;
    d[i] = f2bf(s[(size_t)k * N + n]);
  }
}

// ---------------------------------------------------------------------------
extern "C" void kernel_launch(void* const* d_in, const int* in_sizes, int n_in,
                              void* d_out, int out_size, void* d_ws, size_t ws_size,
                              hipStream_t stream) {
  const float* x     = (const float*)d_in[0];
  const float* amask = (const float*)d_in[1];
  const float* n1g   = (const float*)d_in[2];
  const float* n1b   = (const float*)d_in[3];
  const float* qkvw  = (const float*)d_in[4];
  const float* qkvb  = (const float*)d_in[5];
  const float* projw = (const float*)d_in[6];
  const float* projb = (const float*)d_in[7];
  const float* n2g   = (const float*)d_in[8];
  const float* n2b   = (const float*)d_in[9];
  const float* fc1w  = (const float*)d_in[10];
  const float* fc1b  = (const float*)d_in[11];
  const float* fc2w  = (const float*)d_in[12];
  const float* fc2b  = (const float*)d_in[13];

  // d_out: x[64,139,768] | token_idx[64,138] (int32) | new_mask[64,139,139] | cls[64,196]
  float* out_x    = (float*)d_out;
  int*   out_tok  = (int*)(out_x + 64 * NKEEP1 * CDIM);
  float* out_mask = out_x + 64 * NKEEP1 * CDIM + 64 * NKEEP;
  float* out_cls  = out_mask + 64 * NKEEP1 * NKEEP1;

  char* ws = (char*)d_ws;
  size_t off = 0;
  auto take = [&](size_t bytes) -> char* {
    char* p = ws + off;
    off += (bytes + 255) & ~(size_t)255;
    return p;
  };
  unsigned short* wqT  = (unsigned short*)take((size_t)CDIM * 3 * CDIM * 2);  // [2304][768]
  unsigned short* wpT  = (unsigned short*)take((size_t)CDIM * CDIM * 2);      // [768][768]
  unsigned short* w1T  = (unsigned short*)take((size_t)CDIM * MLPD * 2);      // [3072][768]
  unsigned short* w2T  = (unsigned short*)take((size_t)MLPD * CDIM * 2);      // [768][3072]
  unsigned short* hbuf = (unsigned short*)take((size_t)64 * NP * CDIM * 2);   // 22.0 MB
  unsigned short* qkvs = (unsigned short*)take((size_t)64 * NHEAD * 3 * NP * HD * 2); // 66.1 MB
  float* x1    = (float*)take((size_t)64 * NTOK * CDIM * 4);
  float* clsw  = (float*)take((size_t)64 * NPATCH * 4);
  float* patch = (float*)take((size_t)64 * NPATCH * NPATCH * 4);
  int*   tokw  = (int*)take((size_t)64 * NKEEP * 4);
  float* x2    = (float*)take((size_t)64 * NKEEP1 * CDIM * 4);
  unsigned short* h2 = (unsigned short*)take((size_t)64 * MP * CDIM * 2);
  // Aliases over dead phases: x_att reuses h; MLP intermediate (56.6 MB)
  // overlays contiguous [hbuf .. qkvs] region (88 MB) after proj.
  unsigned short* xatt = hbuf;
  unsigned short* mbuf = hbuf;

  // init accumulators / mask
  fill_kernel<<<(64 * NPATCH + 255) / 256, 256, 0, stream>>>(clsw, 0.f, 64 * NPATCH);
  fill_kernel<<<(64 * NPATCH * NPATCH + 255) / 256, 256, 0, stream>>>(patch, 0.f, 64 * NPATCH * NPATCH);
  fill_kernel<<<(64 * NKEEP1 * NKEEP1 + 255) / 256, 256, 0, stream>>>(out_mask, 1.f, 64 * NKEEP1 * NKEEP1);

  // weights -> bf16, transposed (B^T, K-contiguous per output column)
  cvt_t_kernel<<<(CDIM * 3 * CDIM + 255) / 256, 256, 0, stream>>>(qkvw, wqT, CDIM, 3 * CDIM);
  cvt_t_kernel<<<(CDIM * CDIM + 255) / 256, 256, 0, stream>>>(projw, wpT, CDIM, CDIM);
  cvt_t_kernel<<<(CDIM * MLPD + 255) / 256, 256, 0, stream>>>(fc1w, w1T, CDIM, MLPD);
  cvt_t_kernel<<<(MLPD * CDIM + 255) / 256, 256, 0, stream>>>(fc2w, w2T, MLPD, CDIM);

  // LN1 -> h (bf16, zero-padded to 224 tokens)
  ln1_kernel<<<64 * NP, 256, 0, stream>>>(x, n1g, n1b, hbuf);

  // QKV GEMM: [64*224, 768] x [768, 2304] -> scattered q/k (token-major), v^T
  gemm_bf16_kernel<EPI_QKV><<<dim3(3 * CDIM / 128, 64 * NP / 128), 256, 0, stream>>>(
      hbuf, wqT, qkvb, nullptr, qkvs, 64 * NP, 3 * CDIM, CDIM);

  // attention per (row-tile, b*h)
  attn_kernel<<<dim3(13, 64 * NHEAD), 256, 0, stream>>>(qkvs, amask, xatt, clsw, patch);

  // proj + residual -> x1 f32
  gemm_bf16_kernel<EPI_PROJ><<<dim3(CDIM / 128, 64 * NP / 128), 256, 0, stream>>>(
      xatt, wpT, projb, x, x1, 64 * NP, CDIM, CDIM);

  // token top-k (+ emit cls_attn, token_idx)
  topk_kernel<<<64, 256, 0, stream>>>(clsw, tokw, out_tok, out_cls);

  // gather + LN2 -> x2 f32, h2 bf16
  gather_ln2_kernel<<<64 * MP, 256, 0, stream>>>(x1, tokw, n2g, n2b, x2, h2);

  // attention-pruning mask
  maskk_kernel<<<dim3(NKEEP, 64), 256, 0, stream>>>(patch, tokw, out_mask);

  // MLP: fc1+GELU -> mbuf bf16, fc2 + residual -> out_x
  gemm_bf16_kernel<EPI_FC1><<<dim3(MLPD / 128, 64 * MP / 128), 256, 0, stream>>>(
      h2, w1T, fc1b, nullptr, mbuf, 64 * MP, MLPD, CDIM);
  gemm_bf16_kernel<EPI_FC2><<<dim3(CDIM / 128, 64 * MP / 128), 256, 0, stream>>>(
      mbuf, w2T, fc2b, x2, out_x, 64 * MP, CDIM, MLPD);

  (void)in_sizes; (void)n_in; (void)out_size; (void)ws_size;
}
